// PairwiseDiscriminator_21844203668288
// MI455X (gfx1250) — matrix operand order repaired and checked by reading, not compiled, not run on hardware
//
#include <hip/hip_runtime.h>
#include <hip/hip_bf16.h>
#include <stdint.h>

#define ALPHA 0.2f
#define BB 256
#define TT 2048
#define CC 128
#define NPAIR 8128
#define NLAYER 13
#define RS 40  // LDS chunk row stride in bf16 elements (80 bytes, 16B aligned)

typedef __attribute__((ext_vector_type(16))) __bf16 v16bf;
typedef __attribute__((ext_vector_type(8)))  float  v8f;

union ABv { v16bf v; uint4 q[2]; };

__device__ __forceinline__ unsigned short f2bf(float f) {
  unsigned int u = __float_as_uint(f);
  u += 0x7FFFu + ((u >> 16) & 1u);  // round-to-nearest-even
  return (unsigned short)(u >> 16);
}
__device__ __forceinline__ float bf2f(unsigned short h) {
  return __uint_as_float(((unsigned int)h) << 16);
}

// ---------------- covariance: one block (8 waves) per batch ----------------
__global__ __launch_bounds__(256) void cov_kernel(const float* __restrict__ x,
                                                  float* __restrict__ cov) {
  __shared__ float sred[256];
  __shared__ float smean[CC];
  __shared__ unsigned short schunk[CC * RS];

  const int b = blockIdx.x;
  const int tid = threadIdx.x;
  const float* xb = x + (size_t)b * TT * CC;

  // phase 1: per-channel mean over time (coalesced)
  {
    const int c = tid & (CC - 1);
    const int h = tid >> 7;  // 0 or 1
    float s = 0.f;
    for (int k = h; k < TT; k += 2) s += xb[(size_t)k * CC + c];
    sred[tid] = s;
    __syncthreads();
    if (tid < CC) smean[tid] = (sred[tid] + sred[tid + CC]) * (1.0f / TT);
  }

  const int wave = tid >> 5;
  const int lane = tid & 31;
  const int lh = lane >> 4;
  const int ln = lane & 15;

  v8f acc[8];
#pragma unroll
  for (int t = 0; t < 8; ++t) acc[t] = (v8f){0.f, 0.f, 0.f, 0.f, 0.f, 0.f, 0.f, 0.f};

  for (int k0 = 0; k0 < TT; k0 += 32) {
    __syncthreads();
    // stage centered bf16 chunk [128 channels][32 time] into LDS (float4 reads)
#pragma unroll
    for (int i = 0; i < 4; ++i) {
      int e = i * 256 + tid;       // quad index: 32 k * 32 quads
      int kk = e >> 5;             // 0..31
      int c4 = (e & 31) * 4;       // 0..124
      float4 v = *(const float4*)(xb + (size_t)(k0 + kk) * CC + c4);
      schunk[(c4 + 0) * RS + kk] = f2bf(v.x - smean[c4 + 0]);
      schunk[(c4 + 1) * RS + kk] = f2bf(v.y - smean[c4 + 1]);
      schunk[(c4 + 2) * RS + kk] = f2bf(v.z - smean[c4 + 2]);
      schunk[(c4 + 3) * RS + kk] = f2bf(v.w - smean[c4 + 3]);
    }
    __syncthreads();

    // A tile: rows wave*16..+15 (channels), K = 32 time steps
    ABv a;
    const unsigned short* arow = schunk + (wave * 16 + ln) * RS;
    a.q[0] = *(const uint4*)(arow + 8 * lh);         // K 0..7  (+8 for hi half)
    a.q[1] = *(const uint4*)(arow + 16 + 8 * lh);    // K 16..23 (+8 for hi half)
#pragma unroll
    for (int tn = 0; tn < 8; ++tn) {
      ABv bb;
      const unsigned short* brow = schunk + (tn * 16 + ln) * RS;
      bb.q[0] = *(const uint4*)(brow + 8 * lh);      // K 0..7  (+8 for hi half)
      bb.q[1] = *(const uint4*)(brow + 16 + 8 * lh); // K 16..23 (+8 for hi half)
      acc[tn] = __builtin_amdgcn_wmma_f32_16x16x32_bf16(
          false, a.v, false, bb.v, (short)0, acc[tn], false, false);
    }
  }

  float* covb = cov + (size_t)b * CC * CC;
#pragma unroll
  for (int tn = 0; tn < 8; ++tn) {
#pragma unroll
    for (int r = 0; r < 8; ++r) {
      covb[(size_t)(wave * 16 + r + 8 * lh) * CC + tn * 16 + ln] = acc[tn][r];
    }
  }
}

// ---------------- upper-tri correlation features -> bf16 ----------------
__global__ __launch_bounds__(256) void corr_features_kernel(const float* __restrict__ cov,
                                                            unsigned short* __restrict__ h0) {
  __shared__ float srd[CC];
  const int b = blockIdx.x;
  const int tid = threadIdx.x;
  const float* covb = cov + (size_t)b * CC * CC;
  if (tid < CC) srd[tid] = rsqrtf(fmaxf(covb[tid * CC + tid], 1e-30f));
  __syncthreads();
  for (int p = tid; p < NPAIR; p += 256) {
    // invert p = i*(255-i)/2 + (j-i-1)
    float disc = 65025.0f - 8.0f * (float)p;
    int i = (int)((255.0f - sqrtf(fmaxf(disc, 0.f))) * 0.5f);
    if (i < 0) i = 0;
    if (i > 126) i = 126;
    while (i > 0 && (i * (255 - i)) / 2 > p) --i;
    while (((i + 1) * (254 - i)) / 2 <= p) ++i;
    int j = p - (i * (255 - i)) / 2 + i + 1;
    float corr = covb[i * CC + j] * srd[i] * srd[j];
    corr = fminf(1.0f, fmaxf(-1.0f, corr));
    h0[(size_t)b * NPAIR + p] = f2bf(corr);
  }
}

// ---------------- bf16 GEMM: one wave per 32x32 output strip (2x2 tiles) ---
// Out[M=256][Npad] = act(A[256][Kpad] * W[Npad][Kpad]^T + bias)
__global__ __launch_bounds__(256) void gemm_kernel(const unsigned short* __restrict__ A,
                                                   const unsigned short* __restrict__ W,
                                                   const float* __restrict__ bias,
                                                   unsigned short* __restrict__ Out,
                                                   int Kpad, int Npad, int leaky) {
  const int wave = threadIdx.x >> 5;
  const int lane = threadIdx.x & 31;
  const int N32 = Npad >> 5;
  const int tile = blockIdx.x * 8 + wave;
  if (tile >= (BB / 32) * N32) return;  // wave-uniform exit, EXEC stays full
  const int tm2 = tile / N32;
  const int tn2 = tile - tm2 * N32;
  const int lh = lane >> 4;
  const int ln = lane & 15;

  const unsigned short* arow0 = A + (size_t)(tm2 * 32 + ln) * Kpad;
  const unsigned short* arow1 = arow0 + (size_t)16 * Kpad;
  const unsigned short* wrow0 = W + (size_t)(tn2 * 32 + ln) * Kpad;
  const unsigned short* wrow1 = wrow0 + (size_t)16 * Kpad;

  v8f acc00 = (v8f){0.f, 0.f, 0.f, 0.f, 0.f, 0.f, 0.f, 0.f};
  v8f acc01 = acc00, acc10 = acc00, acc11 = acc00;

  for (int k0 = 0; k0 < Kpad; k0 += 32) {
    ABv a0, a1, b0, b1;
    a0.q[0] = *(const uint4*)(arow0 + k0 + 8 * lh);
    a0.q[1] = *(const uint4*)(arow0 + k0 + 16 + 8 * lh);
    a1.q[0] = *(const uint4*)(arow1 + k0 + 8 * lh);
    a1.q[1] = *(const uint4*)(arow1 + k0 + 16 + 8 * lh);
    b0.q[0] = *(const uint4*)(wrow0 + k0 + 8 * lh);
    b0.q[1] = *(const uint4*)(wrow0 + k0 + 16 + 8 * lh);
    b1.q[0] = *(const uint4*)(wrow1 + k0 + 8 * lh);
    b1.q[1] = *(const uint4*)(wrow1 + k0 + 16 + 8 * lh);
    acc00 = __builtin_amdgcn_wmma_f32_16x16x32_bf16(false, a0.v, false, b0.v,
                                                    (short)0, acc00, false, false);
    acc01 = __builtin_amdgcn_wmma_f32_16x16x32_bf16(false, a0.v, false, b1.v,
                                                    (short)0, acc01, false, false);
    acc10 = __builtin_amdgcn_wmma_f32_16x16x32_bf16(false, a1.v, false, b0.v,
                                                    (short)0, acc10, false, false);
    acc11 = __builtin_amdgcn_wmma_f32_16x16x32_bf16(false, a1.v, false, b1.v,
                                                    (short)0, acc11, false, false);
  }

  const float bn0 = bias[tn2 * 32 + ln];
  const float bn1 = bias[tn2 * 32 + 16 + ln];
#pragma unroll
  for (int r = 0; r < 8; ++r) {
    int row0 = tm2 * 32 + r + 8 * lh;
    int row1 = row0 + 16;
    float v00 = acc00[r] + bn0;
    float v01 = acc01[r] + bn1;
    float v10 = acc10[r] + bn0;
    float v11 = acc11[r] + bn1;
    if (leaky) {
      v00 = (v00 >= 0.f) ? v00 : ALPHA * v00;
      v01 = (v01 >= 0.f) ? v01 : ALPHA * v01;
      v10 = (v10 >= 0.f) ? v10 : ALPHA * v10;
      v11 = (v11 >= 0.f) ? v11 : ALPHA * v11;
    }
    Out[(size_t)row0 * Npad + tn2 * 32 + ln]      = f2bf(v00);
    Out[(size_t)row0 * Npad + tn2 * 32 + 16 + ln] = f2bf(v01);
    Out[(size_t)row1 * Npad + tn2 * 32 + ln]      = f2bf(v10);
    Out[(size_t)row1 * Npad + tn2 * 32 + 16 + ln] = f2bf(v11);
  }
}

// ---------------- weight / bias conversion to zero-padded bf16 ----------------
__global__ void convert_w_kernel(const float* __restrict__ W, unsigned short* __restrict__ Wb,
                                 int dout, int din, int Kpad, long total) {
  for (long idx = (long)blockIdx.x * blockDim.x + threadIdx.x; idx < total;
       idx += (long)gridDim.x * blockDim.x) {
    int r = (int)(idx / Kpad);
    int c = (int)(idx - (long)r * Kpad);
    float v = (r < dout && c < din) ? W[(size_t)r * din + c] : 0.f;
    Wb[idx] = f2bf(v);
  }
}

__global__ void convert_b_kernel(const float* __restrict__ bsrc, float* __restrict__ bp,
                                 int dout, int Npad) {
  int i = blockIdx.x * blockDim.x + threadIdx.x;
  if (i < Npad) bp[i] = (i < dout) ? bsrc[i] : 0.f;
}

__global__ void sigmoid_kernel(const unsigned short* __restrict__ h, float* __restrict__ out) {
  int s = threadIdx.x;  // 256 threads, final layer Npad = 32, column 0 is real
  float v = bf2f(h[(size_t)s * 32]);
  out[s] = 1.0f / (1.0f + expf(-v));
}

static inline int pad32(int x) { return (x + 31) & ~31; }

extern "C" void kernel_launch(void* const* d_in, const int* in_sizes, int n_in,
                              void* d_out, int out_size, void* d_ws, size_t ws_size,
                              hipStream_t stream) {
  static const int DIN[NLAYER]  = {8128, 4064, 2032, 1016, 508, 254, 127, 63, 31, 15, 7, 3, 1};
  static const int DOUT[NLAYER] = {4064, 2032, 1016, 508, 254, 127, 63, 31, 15, 7, 3, 1, 1};

  (void)in_sizes; (void)n_in; (void)out_size; (void)ws_size;

  const float* x = (const float*)d_in[0];
  const float* Ws[NLAYER];
  const float* bs[NLAYER];
  for (int i = 0; i < NLAYER; ++i) {
    Ws[i] = (const float*)d_in[1 + i];
    bs[i] = (const float*)d_in[1 + NLAYER + i];
  }

  char* ws = (char*)d_ws;
  size_t off = 0;
  auto alloc = [&](size_t bytes) {
    size_t o = off;
    off += (bytes + 255) & ~(size_t)255;
    return o;
  };

  float* cov = (float*)(ws + alloc((size_t)BB * CC * CC * 4));           // 16.8 MB
  unsigned short* hA = (unsigned short*)(ws + alloc((size_t)BB * NPAIR * 2));
  unsigned short* hB = (unsigned short*)(ws + alloc((size_t)BB * NPAIR * 2));
  unsigned short* Wb[NLAYER];
  float* bp[NLAYER];
  for (int i = 0; i < NLAYER; ++i) {
    Wb[i] = (unsigned short*)(ws + alloc((size_t)pad32(DOUT[i]) * pad32(DIN[i]) * 2));
    bp[i] = (float*)(ws + alloc((size_t)pad32(DOUT[i]) * 4));
  }

  // convert weights/biases to zero-padded bf16 / f32 (part of the captured graph)
  for (int i = 0; i < NLAYER; ++i) {
    int Kp = pad32(DIN[i]), Np = pad32(DOUT[i]);
    long total = (long)Np * Kp;
    int blocks = (int)((total + 255) / 256);
    if (blocks > 4096) blocks = 4096;
    convert_w_kernel<<<blocks, 256, 0, stream>>>(Ws[i], Wb[i], DOUT[i], DIN[i], Kp, total);
    convert_b_kernel<<<(Np + 255) / 256, 256, 0, stream>>>(bs[i], bp[i], DOUT[i], Np);
  }

  cov_kernel<<<BB, 256, 0, stream>>>(x, cov);
  corr_features_kernel<<<BB, 256, 0, stream>>>(cov, hA);

  unsigned short* cur = hA;
  unsigned short* nxt = hB;
  for (int i = 0; i < NLAYER; ++i) {
    int Kp = pad32(DIN[i]), Np = pad32(DOUT[i]);
    int tiles = (BB / 32) * (Np / 32);
    int blocks = (tiles + 7) / 8;
    gemm_kernel<<<blocks, 256, 0, stream>>>(cur, Wb[i], bp[i], nxt, Kp, Np,
                                            (i < NLAYER - 1) ? 1 : 0);
    unsigned short* t = cur; cur = nxt; nxt = t;
  }

  sigmoid_kernel<<<1, BB, 0, stream>>>(cur, (float*)d_out);
}
